// QuantizedQwen3MoeMLP_39865886442066
// MI455X (gfx1250) — 8-bit matrix operand order repaired and checked by reading, compiled, not run on hardware
//
#include <hip/hip_runtime.h>
#include <cstdint>
#include <cstddef>


typedef __attribute__((ext_vector_type(8))) int v8i;

#define H_DIM 2048
#define I_DIM 5632
#define M_TOK 8192

#define TM 128
#define TN 128
#define TK 64
#define LDST 80

__global__ __launch_bounds__(256) void quant_rows(const float* __restrict__ in,
                                                  int8_t* __restrict__ outq,
                                                  float* __restrict__ scales,
                                                  int K) {
  const int row = blockIdx.x;
  const float* r = in + (size_t)row * K;
  int8_t* o = outq + (size_t)row * K;

  float m = 0.0f;
  for (int i = threadIdx.x; i < K; i += 256) m = fmaxf(m, fabsf(r[i]));

  __shared__ float red[256];
  red[threadIdx.x] = m;
  __syncthreads();
#pragma unroll
  for (int s = 128; s > 0; s >>= 1) {
    if (threadIdx.x < (unsigned)s) red[threadIdx.x] = fmaxf(red[threadIdx.x], red[threadIdx.x + s]);
    __syncthreads();
  }
  const float scale = fmaxf(red[0] * (1.0f / 127.0f), 1e-8f);
  const float inv = 1.0f / scale;
  if (threadIdx.x == 0) scales[row] = scale;

  for (int i = threadIdx.x; i < K; i += 256) {
    float q = rintf(r[i] * inv);
    q = fminf(fmaxf(q, -127.0f), 127.0f);
    o[i] = (int8_t)(int)q;
  }
}

__device__ __forceinline__ void fetch_tile16(const int8_t* __restrict__ src, int rowBase,
                                             int ldK, int k0, int t, int4& r0, int4& r1) {
  const int c0 = t, c1 = t + 256;
  const int8_t* p0 = src + (size_t)(rowBase + (c0 >> 2)) * (size_t)ldK + (size_t)(k0 + (c0 & 3) * 16);
  const int8_t* p1 = src + (size_t)(rowBase + (c1 >> 2)) * (size_t)ldK + (size_t)(k0 + (c1 & 3) * 16);
  r0 = *reinterpret_cast<const int4*>(p0);
  r1 = *reinterpret_cast<const int4*>(p1);
}

__device__ __forceinline__ void store_tile_lds(char* lds, int t, const int4& r0, const int4& r1) {
  const int c0 = t, c1 = t + 256;
  *reinterpret_cast<int4*>(lds + (c0 >> 2) * LDST + (c0 & 3) * 16) = r0;
  *reinterpret_cast<int4*>(lds + (c1 >> 2) * LDST + (c1 & 3) * 16) = r1;
}

__device__ __forceinline__ v8i load_a_frag(const char* lds, int fragRow, int lane) {
  const int r = lane & 15, half = lane >> 4;
  const char* p = lds + (fragRow + r) * LDST + half * 8;
  int2 q0 = *reinterpret_cast<const int2*>(p + 0);
  int2 q1 = *reinterpret_cast<const int2*>(p + 16);
  int2 q2 = *reinterpret_cast<const int2*>(p + 32);
  int2 q3 = *reinterpret_cast<const int2*>(p + 48);
  v8i a;
  a[0] = q0.x; a[1] = q0.y; a[2] = q1.x; a[3] = q1.y;
  a[4] = q2.x; a[5] = q2.y; a[6] = q3.x; a[7] = q3.y;
  return a;
}

__device__ __forceinline__ v8i load_b_frag(const char* lds, int fragRow, int lane) {
  const int c = lane & 15, half = lane >> 4;
  const char* p = lds + (fragRow + c) * LDST + half * 8;
  int2 q0 = *reinterpret_cast<const int2*>(p + 0);
  int2 q1 = *reinterpret_cast<const int2*>(p + 16);
  int2 q2 = *reinterpret_cast<const int2*>(p + 32);
  int2 q3 = *reinterpret_cast<const int2*>(p + 48);
  v8i b;
  b[0] = q0.x; b[1] = q0.y; b[2] = q1.x; b[3] = q1.y;
  b[4] = q2.x; b[5] = q2.y; b[6] = q3.x; b[7] = q3.y;
  return b;
}

#define WMMA_IU8(A, B, C) \
  __builtin_amdgcn_wmma_i32_16x16x64_iu8(true, (A), true, (B), (C), false, false)

__global__ __launch_bounds__(256, 1) void gemm1_silu_int8(
    const int8_t* __restrict__ xq, const float* __restrict__ sx,
    const int8_t* __restrict__ w1q, const float* __restrict__ sw1,
    float* __restrict__ h) {
  __shared__ char ldsA[TM * LDST];
  __shared__ char ldsG[TN * LDST];
  __shared__ char ldsU[TN * LDST];
  __shared__ float sxT[TM], sgT[TN], suT[TN];

  const int t = threadIdx.x;
  const int lane = t & 31;
  const int wave = t >> 5;
  const int wm = wave >> 1;
  const int wn = wave & 1;
  const int mBase = blockIdx.y * TM;
  const int nBase = blockIdx.x * TN;

  if (t < TM) {
    sxT[t] = sx[mBase + t];
    sgT[t] = sw1[nBase + t];
    suT[t] = sw1[I_DIM + nBase + t];
  }

  v8i accG[2][4], accU[2][4];
#pragma unroll
  for (int i = 0; i < 2; ++i)
#pragma unroll
    for (int j = 0; j < 4; ++j) {
      accG[i][j] = (v8i){0, 0, 0, 0, 0, 0, 0, 0};
      accU[i][j] = (v8i){0, 0, 0, 0, 0, 0, 0, 0};
    }

  int4 a0, a1, g0, g1, u0, u1;
  fetch_tile16(xq, mBase, H_DIM, 0, t, a0, a1);
  fetch_tile16(w1q, nBase, H_DIM, 0, t, g0, g1);
  fetch_tile16(w1q, I_DIM + nBase, H_DIM, 0, t, u0, u1);

  const int KB = H_DIM / TK;
  for (int kb = 0; kb < KB; ++kb) {
    __syncthreads();
    store_tile_lds(ldsA, t, a0, a1);
    store_tile_lds(ldsG, t, g0, g1);
    store_tile_lds(ldsU, t, u0, u1);
    __syncthreads();
    if (kb + 1 < KB) {
      const int k0 = (kb + 1) * TK;
      fetch_tile16(xq, mBase, H_DIM, k0, t, a0, a1);
      fetch_tile16(w1q, nBase, H_DIM, k0, t, g0, g1);
      fetch_tile16(w1q, I_DIM + nBase, H_DIM, k0, t, u0, u1);
    }
    const v8i af0 = load_a_frag(ldsA, wm * 32 + 0, lane);
    const v8i af1 = load_a_frag(ldsA, wm * 32 + 16, lane);
#pragma unroll
    for (int fn = 0; fn < 4; ++fn) {
      const v8i bg = load_b_frag(ldsG, wn * 64 + fn * 16, lane);
      accG[0][fn] = WMMA_IU8(af0, bg, accG[0][fn]);
      accG[1][fn] = WMMA_IU8(af1, bg, accG[1][fn]);
      const v8i bu = load_b_frag(ldsU, wn * 64 + fn * 16, lane);
      accU[0][fn] = WMMA_IU8(af0, bu, accU[0][fn]);
      accU[1][fn] = WMMA_IU8(af1, bu, accU[1][fn]);
    }
  }

  const int half = lane >> 4, cl = lane & 15;
#pragma unroll
  for (int fm = 0; fm < 2; ++fm)
#pragma unroll
    for (int fn = 0; fn < 4; ++fn)
#pragma unroll
      for (int j = 0; j < 8; ++j) {
        const int ml = wm * 32 + fm * 16 + half * 8 + j;
        const int nl = wn * 64 + fn * 16 + cl;
        const float sxv = sxT[ml];
        const float g = (float)accG[fm][fn][j] * sxv * sgT[nl];
        const float u = (float)accU[fm][fn][j] * sxv * suT[nl];
        const float act = g / (1.0f + __expf(-g));
        h[(size_t)(mBase + ml) * I_DIM + (size_t)(nBase + nl)] = act * u;
      }
}

__global__ __launch_bounds__(256, 1) void gemm2_int8(
    const int8_t* __restrict__ aq, const float* __restrict__ sa,
    const int8_t* __restrict__ bq, const float* __restrict__ sb,
    float* __restrict__ out) {
  __shared__ char ldsA[TM * LDST];
  __shared__ char ldsB[TN * LDST];
  __shared__ float saT[TM], sbT[TN];

  const int t = threadIdx.x;
  const int lane = t & 31;
  const int wave = t >> 5;
  const int wm = wave >> 1;
  const int wn = wave & 1;
  const int mBase = blockIdx.y * TM;
  const int nBase = blockIdx.x * TN;

  if (t < TM) {
    saT[t] = sa[mBase + t];
    sbT[t] = sb[nBase + t];
  }

  v8i acc[2][4];
#pragma unroll
  for (int i = 0; i < 2; ++i)
#pragma unroll
    for (int j = 0; j < 4; ++j) acc[i][j] = (v8i){0, 0, 0, 0, 0, 0, 0, 0};

  int4 a0, a1, b0, b1;
  fetch_tile16(aq, mBase, I_DIM, 0, t, a0, a1);
  fetch_tile16(bq, nBase, I_DIM, 0, t, b0, b1);

  const int KB = I_DIM / TK;
  for (int kb = 0; kb < KB; ++kb) {
    __syncthreads();
    store_tile_lds(ldsA, t, a0, a1);
    store_tile_lds(ldsB, t, b0, b1);
    __syncthreads();
    if (kb + 1 < KB) {
      const int k0 = (kb + 1) * TK;
      fetch_tile16(aq, mBase, I_DIM, k0, t, a0, a1);
      fetch_tile16(bq, nBase, I_DIM, k0, t, b0, b1);
    }
    const v8i af0 = load_a_frag(ldsA, wm * 32 + 0, lane);
    const v8i af1 = load_a_frag(ldsA, wm * 32 + 16, lane);
#pragma unroll
    for (int fn = 0; fn < 4; ++fn) {
      const v8i bf = load_b_frag(ldsB, wn * 64 + fn * 16, lane);
      acc[0][fn] = WMMA_IU8(af0, bf, acc[0][fn]);
      acc[1][fn] = WMMA_IU8(af1, bf, acc[1][fn]);
    }
  }

  const int half = lane >> 4, cl = lane & 15;
#pragma unroll
  for (int fm = 0; fm < 2; ++fm)
#pragma unroll
    for (int fn = 0; fn < 4; ++fn)
#pragma unroll
      for (int j = 0; j < 8; ++j) {
        const int ml = wm * 32 + fm * 16 + half * 8 + j;
        const int nl = wn * 64 + fn * 16 + cl;
        const float v = (float)acc[fm][fn][j] * saT[ml] * sbT[nl];
        out[(size_t)(mBase + ml) * H_DIM + (size_t)(nBase + nl)] = v;
      }
}

extern "C" void kernel_launch(void* const* d_in, const int* in_sizes, int n_in,
                              void* d_out, int out_size, void* d_ws, size_t ws_size,
                              hipStream_t stream) {
  (void)in_sizes; (void)n_in; (void)out_size; (void)ws_size;
  const float* x  = (const float*)d_in[0];
  const float* w1 = (const float*)d_in[1];
  const float* w2 = (const float*)d_in[2];
  float* out = (float*)d_out;

  char* ws = (char*)d_ws;
  size_t off = 0;
  auto take = [&](size_t n) -> void* {
    void* p = (void*)(ws + off);
    off += (n + 255) & ~(size_t)255;
    return p;
  };
  int8_t* xq  = (int8_t*)take((size_t)M_TOK * H_DIM);
  float*  sx  = (float*)take((size_t)M_TOK * sizeof(float));
  int8_t* w1q = (int8_t*)take((size_t)2 * I_DIM * H_DIM);
  float*  sw1 = (float*)take((size_t)2 * I_DIM * sizeof(float));
  int8_t* w2q = (int8_t*)take((size_t)H_DIM * I_DIM);
  float*  sw2 = (float*)take((size_t)H_DIM * sizeof(float));
  float*  h   = (float*)take((size_t)M_TOK * I_DIM * sizeof(float));
  int8_t* hq  = (int8_t*)take((size_t)M_TOK * I_DIM);
  float*  sh  = (float*)take((size_t)M_TOK * sizeof(float));

  quant_rows<<<M_TOK, 256, 0, stream>>>(x, xq, sx, H_DIM);
  quant_rows<<<2 * I_DIM, 256, 0, stream>>>(w1, w1q, sw1, H_DIM);
  quant_rows<<<H_DIM, 256, 0, stream>>>(w2, w2q, sw2, I_DIM);

  gemm1_silu_int8<<<dim3(I_DIM / TN, M_TOK / TM), 256, 0, stream>>>(xq, sx, w1q, sw1, h);

  quant_rows<<<M_TOK, 256, 0, stream>>>(h, hq, sh, I_DIM);

  gemm2_int8<<<dim3(H_DIM / TN, M_TOK / TM), 256, 0, stream>>>(hq, sh, w2q, sw2, out);
}
